// GotenInteractionLayer_25177098289384
// MI455X (gfx1250) — matrix operand order repaired and checked by reading, compiled, not run on hardware
//
#include <hip/hip_runtime.h>
#include <hip/hip_bf16.h>
#include <math.h>

#define N_NODESC 5000
#define N_EDGESC 20000
#define LATC 32
#define MULTC 32
#define SPHC 9
#define GEN_NC 3200
#define NTILES_ENV 200   // packed tiles; only 0..197 are consumed
#define EB 16

typedef __attribute__((ext_vector_type(16))) __bf16 bf16x16;
typedef __attribute__((ext_vector_type(8)))  float  v8f;
typedef __attribute__((ext_vector_type(4)))  unsigned int u32x4;

union Frag {
  bf16x16 v;
  unsigned short s[16];
  u32x4 q[2];
};

__device__ __forceinline__ unsigned short f2bf(float f) {
  union { float f; unsigned u; } v; v.f = f;
  unsigned u = v.u;
  u += 0x7fffu + ((u >> 16) & 1u);
  return (unsigned short)(u >> 16);
}
__device__ __forceinline__ float siluf(float x) { return x / (1.0f + expf(-x)); }
__device__ __forceinline__ unsigned enc_ord(float f) {
  union { float f; unsigned u; } v; v.f = f;
  return (v.u & 0x80000000u) ? ~v.u : (v.u | 0x80000000u);
}
__device__ __forceinline__ float dec_ord(unsigned e) {
  union { float f; unsigned u; } v;
  v.u = (e & 0x80000000u) ? (e & 0x7fffffffu) : ~e;
  return v.f;
}
__device__ __forceinline__ v8f wmma_bf16(const Frag& a, const Frag& b, v8f c) {
  return __builtin_amdgcn_wmma_f32_16x16x32_bf16(false, a.v, false, b.v,
                                                 (short)0, c, false, false);
}

// Pack W[k][col] (row-major KxN, K=32) into per-tile WMMA B fragments:
// frag[tile][lane][e] = bf16(W[(e<8 ? (lane>>4)*8+e : 16+(lane>>4)*8+e-8)][tile*16+(lane&15)])
__global__ void k_pack(const float* __restrict__ src, unsigned short* __restrict__ dst,
                       int ncols, int ntiles) {
  int idx = blockIdx.x * blockDim.x + threadIdx.x;
  if (idx >= ntiles * 512) return;
  int tile = idx >> 9;
  int lane = (idx >> 4) & 31;
  int e = idx & 15;
  int col = tile * 16 + (lane & 15);
  int k = (e < 8) ? (lane >> 4) * 8 + e : 16 + (lane >> 4) * 8 + (e - 8);
  dst[idx] = f2bf(src[(size_t)k * ncols + col]);
}

// Node-side irreps_linear: out[n,j,c] = (1/sqrt(32)) * sum_i X[n,i,c]*W[l(c)][i][j]
__global__ void k_irreps(const float* __restrict__ X, const float* __restrict__ W,
                         float* __restrict__ out) {
  int idx = blockIdx.x * blockDim.x + threadIdx.x;
  if (idx >= N_NODESC * MULTC * SPHC) return;
  int c = idx % SPHC;
  int j = (idx / SPHC) % MULTC;
  int n = idx / (SPHC * MULTC);
  int l = (c < 1) ? 0 : ((c < 4) ? 1 : 2);
  const float* w = W + l * MULTC * MULTC;
  const float* x = X + (size_t)n * MULTC * SPHC + c;
  float s = 0.f;
  #pragma unroll
  for (int i = 0; i < MULTC; ++i) s += x[i * SPHC] * w[i * MULTC + j];
  out[idx] = s * 0.17677669529663689f;
}

// Fused env kernel: env = (t@W_rs) * (silu(h_j@W1+b1)@W2 + b2), fused consumers.
// Head tiles (cols 0..95): delta_h atomics + spharm outer product.
// Lin tiles (cols 96..3167): per-edge irreps_linear, register-accumulated over i.
// Cols 3168..3199 are never consumed by the reference and are skipped.
__global__ void __launch_bounds__(256) k_env(
    const float* __restrict__ t_ij, const float* __restrict__ h,
    const float* __restrict__ X, const int* __restrict__ ecenter,
    const int* __restrict__ eneigh, const float* __restrict__ spharms,
    const float* __restrict__ hj_W1, const float* __restrict__ hj_b1,
    const float* __restrict__ hj_b2,
    const unsigned short* __restrict__ wrs_frag,
    const unsigned short* __restrict__ w2_frag,
    float* __restrict__ dh_accum, float* __restrict__ dXij) {
  __shared__ int ec[EB], en[EB];
  __shared__ float hj[EB][LATC];
  __shared__ float w1s[LATC * LATC];
  __shared__ unsigned short t_bf[EB][LATC];
  __shared__ unsigned short hid_bf[EB][LATC];
  __shared__ float sph[EB][SPHC];
  __shared__ float Xj[EB][MULTC][SPHC];
  __shared__ float acc[EB][MULTC][SPHC];

  const int tid = threadIdx.x;
  const int e0 = blockIdx.x * EB;
  if (tid < EB) { ec[tid] = ecenter[e0 + tid]; en[tid] = eneigh[e0 + tid]; }
  __syncthreads();

  for (int i = tid; i < EB * LATC; i += 256) {
    int e = i >> 5, k = i & 31;
    t_bf[e][k] = f2bf(t_ij[(size_t)(e0 + e) * LATC + k]);
    hj[e][k] = h[(size_t)en[e] * LATC + k];
  }
  for (int i = tid; i < LATC * LATC; i += 256) w1s[i] = hj_W1[i];
  for (int i = tid; i < EB * SPHC; i += 256)
    sph[i / SPHC][i % SPHC] = spharms[(size_t)(e0 + i / SPHC) * SPHC + i % SPHC];
  for (int i = tid; i < EB * MULTC * SPHC; i += 256) {
    int e = i / (MULTC * SPHC), r = i % (MULTC * SPHC);
    ((float*)Xj)[i] = X[(size_t)en[e] * MULTC * SPHC + r];
    ((float*)acc)[i] = 0.f;
  }
  __syncthreads();

  for (int i = tid; i < EB * LATC; i += 256) {
    int e = i >> 5, j = i & 31;
    float s = hj_b1[j];
    #pragma unroll
    for (int k = 0; k < LATC; ++k) s += hj[e][k] * w1s[k * LATC + j];
    hid_bf[e][j] = f2bf(siluf(s));
  }
  __syncthreads();

  const int lane = tid & 31;
  const int wave = tid >> 5;
  const int row = lane & 15;
  const int g = lane >> 4;
  const int mbase = g * 8;

  Frag aT, aH;
  #pragma unroll
  for (int e2 = 0; e2 < 8; ++e2) {
    aT.s[e2]     = t_bf[row][g * 8 + e2];
    aT.s[8 + e2] = t_bf[row][16 + g * 8 + e2];
    aH.s[e2]     = hid_bf[row][g * 8 + e2];
    aH.s[8 + e2] = hid_bf[row][16 + g * 8 + e2];
  }

  const int LOFF[3] = {0, 1, 4};
  const int LDIM[3] = {1, 3, 5};
  const float scale = 0.17677669529663689f;

  // ---- head tiles nt = 0..5 (waves 0..5 take one each) ----
  for (int nt = wave; nt < 6; nt += 8) {
    Frag bR, bW;
    const u32x4* pr = (const u32x4*)(wrs_frag + ((size_t)nt * 32 + lane) * 16);
    bR.q[0] = pr[0]; bR.q[1] = pr[1];
    const u32x4* pw = (const u32x4*)(w2_frag + ((size_t)nt * 32 + lane) * 16);
    bW.q[0] = pw[0]; bW.q[1] = pw[1];
    float bias = hj_b2[nt * 16 + row];
    v8f c1 = {0.f, 0.f, 0.f, 0.f, 0.f, 0.f, 0.f, 0.f};
    v8f c2 = {0.f, 0.f, 0.f, 0.f, 0.f, 0.f, 0.f, 0.f};
    c1 = wmma_bf16(aT, bR, c1);
    c2 = wmma_bf16(aH, bW, c2);
    float env[8];
    #pragma unroll
    for (int r = 0; r < 8; ++r) env[r] = c1[r] * (c2[r] + bias);

    int col = nt * 16 + row;
    if (nt < 2) {  // delta_h_j columns 0..31 -> segment sum via atomics
      #pragma unroll
      for (int r = 0; r < 8; ++r)
        atomicAdd(&dh_accum[(size_t)ec[mbase + r] * LATC + col], env[r]);
    }
    // ew_w columns 0..95 -> spharm outer product
    int l = col >> 5, mm = col & 31;
    int o = LOFF[l], d = LDIM[l];
    for (int r = 0; r < 8; ++r) {
      int e = mbase + r;
      float wv = env[r];
      for (int c3 = 0; c3 < d; ++c3)
        atomicAdd(&acc[e][mm][o + c3], wv * sph[e][o + c3]);
    }
  }

  // ---- lin tiles: p = l*64 + jhalf*32 + i, contiguous run of 24 per wave ----
  float accR[8][5];
  #pragma unroll
  for (int r = 0; r < 8; ++r)
    #pragma unroll
    for (int c3 = 0; c3 < 5; ++c3) accR[r][c3] = 0.f;

  const int pbeg = wave * 24, pend = pbeg + 24;
  for (int p = pbeg; p < pend; ++p) {
    int l = p >> 6;
    int jh = (p >> 5) & 1;
    int i2 = p & 31;
    int nt = 6 + l * 64 + i2 * 2 + jh;
    const unsigned short* fr = wrs_frag + ((size_t)nt * 32 + lane) * 16;
    const unsigned short* fw = w2_frag + ((size_t)nt * 32 + lane) * 16;
    if (p + 1 < pend) {  // prefetch next tile's fragments (global_prefetch_b8)
      int pn = p + 1;
      int ntn = 6 + (pn >> 6) * 64 + (pn & 31) * 2 + ((pn >> 5) & 1);
      __builtin_prefetch(wrs_frag + ((size_t)ntn * 32 + lane) * 16, 0, 3);
      __builtin_prefetch(w2_frag + ((size_t)ntn * 32 + lane) * 16, 0, 3);
    }
    Frag bR, bW;
    bR.q[0] = ((const u32x4*)fr)[0]; bR.q[1] = ((const u32x4*)fr)[1];
    bW.q[0] = ((const u32x4*)fw)[0]; bW.q[1] = ((const u32x4*)fw)[1];
    float bias = hj_b2[nt * 16 + row];
    v8f c1 = {0.f, 0.f, 0.f, 0.f, 0.f, 0.f, 0.f, 0.f};
    v8f c2 = {0.f, 0.f, 0.f, 0.f, 0.f, 0.f, 0.f, 0.f};
    c1 = wmma_bf16(aT, bR, c1);
    c2 = wmma_bf16(aH, bW, c2);

    int o = LOFF[l], d = LDIM[l];
    #pragma unroll
    for (int r = 0; r < 8; ++r) {
      float env = c1[r] * (c2[r] + bias);
      for (int c3 = 0; c3 < d; ++c3)
        accR[r][c3] += env * Xj[mbase + r][i2][o + c3];
    }
    // flush at (l,jhalf) group boundary or end of run
    if ((((p + 1) & 31) == 0) || (p + 1 == pend)) {
      int j = jh * 16 + row;
      #pragma unroll
      for (int r = 0; r < 8; ++r)
        for (int c3 = 0; c3 < d; ++c3) {
          atomicAdd(&acc[mbase + r][j][o + c3], scale * accR[r][c3]);
          accR[r][c3] = 0.f;
        }
    }
  }
  __syncthreads();
  for (int i = tid; i < EB * MULTC * SPHC; i += 256)
    dXij[(size_t)e0 * MULTC * SPHC + i] = ((float*)acc)[i];
}

// Per-node layernorm of segment-summed delta_h; h1 = h + ln(delta_h)
__global__ void k_dhln(const float* __restrict__ dh, const float* __restrict__ h,
                       const float* __restrict__ g, const float* __restrict__ b,
                       float* __restrict__ h1) {
  int n = blockIdx.x, lane = threadIdx.x;
  float v = dh[(size_t)n * LATC + lane];
  float s = v;
  for (int off = 16; off >= 1; off >>= 1) s += __shfl_xor(s, off, 32);
  float mu = s * (1.0f / 32.0f);
  float d = v - mu;
  float s2 = d * d;
  for (int off = 16; off >= 1; off >>= 1) s2 += __shfl_xor(s2, off, 32);
  float out = d * rsqrtf(s2 * (1.0f / 32.0f) + 1e-5f) * g[lane] + b[lane];
  h1[(size_t)n * LATC + lane] = h[(size_t)n * LATC + lane] + out;
}

// Edge kernel: tps -> so3 norm -> gated delta_t -> t_new; Q/K WMMA -> logits + segmax
__global__ void __launch_bounds__(256) k_edge_attn(
    const float* __restrict__ t_ij, const float* __restrict__ h,
    const int* __restrict__ ecenter, const int* __restrict__ eneigh,
    const float* __restrict__ XQ, const float* __restrict__ XK,
    const float* __restrict__ gw_W1, const float* __restrict__ gw_b1,
    const float* __restrict__ gw_W2, const float* __restrict__ gw_b2,
    const float* __restrict__ gt_W1, const float* __restrict__ gt_b1,
    const float* __restrict__ gt_W2, const float* __restrict__ gt_b2,
    const float* __restrict__ uc,
    const unsigned short* __restrict__ wq_frag,
    const unsigned short* __restrict__ wk_frag,
    float* __restrict__ t_out, float* __restrict__ logits,
    unsigned* __restrict__ segmax) {
  __shared__ int ec[EB], en[EB];
  __shared__ float tloc[EB][LATC];
  __shared__ float tps[EB][MULTC][3];
  __shared__ float rmsv[EB][3];
  __shared__ float wij[EB][96];
  __shared__ float hidw[EB][LATC];
  __shared__ float hidt[EB][LATC];
  __shared__ unsigned short tn_bf[EB][LATC];
  __shared__ unsigned short h_bf[EB][LATC];
  __shared__ float gw1s[96 * 32];
  __shared__ float gw2s[32 * 32];
  __shared__ float gt1s[32 * 32];
  __shared__ float gt2s[32 * 32];

  const int tid = threadIdx.x;
  const int e0 = blockIdx.x * EB;
  if (tid < EB) { ec[tid] = ecenter[e0 + tid]; en[tid] = eneigh[e0 + tid]; }
  for (int i = tid; i < 96 * 32; i += 256) gw1s[i] = gw_W1[i];
  for (int i = tid; i < 32 * 32; i += 256) {
    gw2s[i] = gw_W2[i]; gt1s[i] = gt_W1[i]; gt2s[i] = gt_W2[i];
  }
  __syncthreads();
  for (int i = tid; i < EB * LATC; i += 256) {
    int e = i >> 5, k = i & 31;
    tloc[e][k] = t_ij[(size_t)(e0 + e) * LATC + k];
    h_bf[e][k] = f2bf(h[(size_t)en[e] * LATC + k]);
  }
  __syncthreads();

  const int LOFF[3] = {0, 1, 4};
  const int LDIM[3] = {1, 3, 5};
  const float ISD[3] = {1.0f, 0.5773502691896258f, 0.4472135954999579f};
  for (int i = tid; i < EB * MULTC; i += 256) {
    int e = i >> 5, m = i & 31;
    const float* q = XQ + (size_t)ec[e] * MULTC * SPHC + m * SPHC;
    const float* kk = XK + (size_t)en[e] * MULTC * SPHC + m * SPHC;
    for (int l = 0; l < 3; ++l) {
      float s = 0.f;
      for (int c = 0; c < LDIM[l]; ++c) s += q[LOFF[l] + c] * kk[LOFF[l] + c];
      tps[e][m][l] = s * ISD[l];
    }
  }
  __syncthreads();
  if (tid < EB * 3) {
    int e = tid / 3, l = tid % 3;
    float s = 0.f;
    for (int m = 0; m < MULTC; ++m) { float v = tps[e][m][l]; s += v * v; }
    rmsv[e][l] = rsqrtf(s * (1.0f / 32.0f) + 1e-8f);
  }
  __syncthreads();
  for (int i = tid; i < EB * MULTC * 3; i += 256) {
    int e = i / 96, r = i % 96;
    wij[e][r] = tps[e][r / 3][r % 3] * rmsv[e][r % 3];
  }
  __syncthreads();
  for (int i = tid; i < EB * LATC; i += 256) {
    int e = i >> 5, j = i & 31;
    float s = gw_b1[j];
    for (int k = 0; k < 96; ++k) s += wij[e][k] * gw1s[k * 32 + j];
    hidw[e][j] = siluf(s);
    float s2 = gt_b1[j];
    #pragma unroll
    for (int k = 0; k < 32; ++k) s2 += tloc[e][k] * gt1s[k * 32 + j];
    hidt[e][j] = siluf(s2);
  }
  __syncthreads();
  float u = uc[0];
  float c0 = rsqrtf(u * u + 1.0f);
  for (int i = tid; i < EB * LATC; i += 256) {
    int e = i >> 5, j = i & 31;
    float a = gw_b2[j], b = gt_b2[j];
    #pragma unroll
    for (int k = 0; k < 32; ++k) {
      a += hidw[e][k] * gw2s[k * 32 + j];
      b += hidt[e][k] * gt2s[k * 32 + j];
    }
    float tn = c0 * tloc[e][j] + u * c0 * (a * b);
    t_out[(size_t)(e0 + e) * LATC + j] = tn;
    tn_bf[e][j] = f2bf(tn);
  }
  __syncthreads();

  const int lane = tid & 31, wave = tid >> 5;
  const int row = lane & 15, g2 = lane >> 4;
  Frag aQ, aK;
  #pragma unroll
  for (int e2 = 0; e2 < 8; ++e2) {
    aQ.s[e2]     = tn_bf[row][g2 * 8 + e2];
    aQ.s[8 + e2] = tn_bf[row][16 + g2 * 8 + e2];
    aK.s[e2]     = h_bf[row][g2 * 8 + e2];
    aK.s[8 + e2] = h_bf[row][16 + g2 * 8 + e2];
  }
  for (int m = wave; m < MULTC; m += 8) {
    if (m + 8 < MULTC) {
      __builtin_prefetch(wq_frag + ((size_t)(m + 8) * 32 + lane) * 16, 0, 3);
      __builtin_prefetch(wk_frag + ((size_t)(m + 8) * 32 + lane) * 16, 0, 3);
    }
    Frag bQ, bK;
    const u32x4* pq = (const u32x4*)(wq_frag + ((size_t)m * 32 + lane) * 16);
    bQ.q[0] = pq[0]; bQ.q[1] = pq[1];
    const u32x4* pk = (const u32x4*)(wk_frag + ((size_t)m * 32 + lane) * 16);
    bK.q[0] = pk[0]; bK.q[1] = pk[1];
    v8f cq = {0.f, 0.f, 0.f, 0.f, 0.f, 0.f, 0.f, 0.f};
    v8f ck = {0.f, 0.f, 0.f, 0.f, 0.f, 0.f, 0.f, 0.f};
    cq = wmma_bf16(aQ, bQ, cq);
    ck = wmma_bf16(aK, bK, ck);
    float p[8];
    #pragma unroll
    for (int r = 0; r < 8; ++r) p[r] = cq[r] * ck[r];
    #pragma unroll
    for (int off = 1; off <= 8; off <<= 1) {
      #pragma unroll
      for (int r = 0; r < 8; ++r) p[r] += __shfl_xor(p[r], off, 32);
    }
    if (row == 0) {
      #pragma unroll
      for (int r = 0; r < 8; ++r) {
        int e = g2 * 8 + r;
        float lg = p[r] * 4.0f;  // * isqrt(HEAD) == 4 per reference
        logits[(size_t)(e0 + e) * MULTC + m] = lg;
        atomicMax(&segmax[(size_t)ec[e] * MULTC + m], enc_ord(lg));
      }
    }
  }
}

__global__ void k_softmax1(const float* __restrict__ logits, const int* __restrict__ ecenter,
                           const unsigned* __restrict__ segmax, float* __restrict__ ew,
                           float* __restrict__ denom) {
  int i = blockIdx.x * blockDim.x + threadIdx.x;
  if (i >= N_EDGESC * MULTC) return;
  int e = i >> 5, m = i & 31;
  float M = dec_ord(segmax[(size_t)ecenter[e] * MULTC + m]);
  float w = expf(logits[i] - M);
  ew[i] = w;
  atomicAdd(&denom[(size_t)ecenter[e] * MULTC + m], w);
}

__global__ void k_scatterX(const float* __restrict__ ew, const float* __restrict__ denom,
                           const int* __restrict__ ecenter, const float* __restrict__ dXij,
                           float* __restrict__ dXacc) {
  int i = blockIdx.x * blockDim.x + threadIdx.x;
  if (i >= N_EDGESC * MULTC) return;
  int e = i >> 5, m = i & 31;
  int c = ecenter[e];
  float attn = ew[i] / denom[(size_t)c * MULTC + m];
  const float* src = dXij + (size_t)i * SPHC;
  float* dst = dXacc + ((size_t)c * MULTC + m) * SPHC;
  #pragma unroll
  for (int ch = 0; ch < SPHC; ++ch) atomicAdd(&dst[ch], src[ch] * attn);
}

__global__ void k_node_final(const float* __restrict__ X, const float* __restrict__ h1,
                             const float* __restrict__ dXacc, const float* __restrict__ vu_W,
                             const float* __restrict__ gm_W1, const float* __restrict__ gm_b1,
                             const float* __restrict__ gm_W2, const float* __restrict__ gm_b2,
                             const float* __restrict__ uc,
                             float* __restrict__ h_out, float* __restrict__ X_out) {
  __shared__ float Xs[MULTC][SPHC];
  __shared__ float inbuf[64];
  __shared__ float hid[32];
  int n = blockIdx.x, lane = threadIdx.x;
  const int LOFF[3] = {0, 1, 4};
  const int LDIM[3] = {1, 3, 5};
  float a[SPHC];
  const float* src = dXacc + ((size_t)n * MULTC + lane) * SPHC;
  #pragma unroll
  for (int ch = 0; ch < SPHC; ++ch) a[ch] = src[ch];
  float u = uc[0], c0 = rsqrtf(u * u + 1.0f);
  float Xn[SPHC];
  const float* xin = X + ((size_t)n * MULTC + lane) * SPHC;
  #pragma unroll
  for (int l = 0; l < 3; ++l) {
    float s = 0.f;
    for (int c = 0; c < LDIM[l]; ++c) { float v = a[LOFF[l] + c]; s += v * v; }
    for (int off = 16; off >= 1; off >>= 1) s += __shfl_xor(s, off, 32);
    float inv = rsqrtf(s / (32.0f * LDIM[l]) + 1e-8f);
    for (int c = 0; c < LDIM[l]; ++c) {
      int ch = LOFF[l] + c;
      Xn[ch] = c0 * xin[ch] + u * c0 * (a[ch] * inv);
    }
  }
  #pragma unroll
  for (int ch = 0; ch < SPHC; ++ch) Xs[lane][ch] = Xn[ch];
  __syncthreads();
  float Xp[SPHC];
  #pragma unroll
  for (int ch = 0; ch < SPHC; ++ch) Xp[ch] = 0.f;
  for (int m = 0; m < MULTC; ++m) {
    float w = vu_W[m * LATC + lane];
    #pragma unroll
    for (int ch = 0; ch < SPHC; ++ch) Xp[ch] += Xs[m][ch] * w;
  }
  float s2 = 0.f;
  #pragma unroll
  for (int ch = 0; ch < SPHC; ++ch) s2 += Xp[ch] * Xp[ch];
  float Xpn = sqrtf(s2 + 1e-8f);
  float h1v = h1[(size_t)n * LATC + lane];
  inbuf[lane] = h1v;
  inbuf[32 + lane] = Xpn;
  __syncthreads();
  float s3 = gm_b1[lane];
  #pragma unroll
  for (int k = 0; k < 64; ++k) s3 += inbuf[k] * gm_W1[k * LATC + lane];
  hid[lane] = siluf(s3);
  __syncthreads();
  float m1 = gm_b2[lane], m2 = gm_b2[32 + lane];
  #pragma unroll
  for (int k = 0; k < 32; ++k) {
    float hv = hid[k];
    m1 += hv * gm_W2[k * 64 + lane];
    m2 += hv * gm_W2[k * 64 + 32 + lane];
  }
  h_out[(size_t)n * LATC + lane] = h1v + m1;
  float* xo = X_out + ((size_t)n * MULTC + lane) * SPHC;
  #pragma unroll
  for (int ch = 0; ch < SPHC; ++ch) xo[ch] = Xn[ch] + m2 * Xp[ch];
}

extern "C" void kernel_launch(void* const* d_in, const int* in_sizes, int n_in,
                              void* d_out, int out_size, void* d_ws, size_t ws_size,
                              hipStream_t stream) {
  (void)in_sizes; (void)n_in; (void)out_size; (void)ws_size;
  const float* h       = (const float*)d_in[0];
  const float* X       = (const float*)d_in[1];
  const float* t_ij    = (const float*)d_in[2];
  const int*   ec      = (const int*)d_in[3];
  const int*   en      = (const int*)d_in[4];
  const float* sph     = (const float*)d_in[6];
  const float* uc      = (const float*)d_in[7];
  const float* W_rs    = (const float*)d_in[9];
  const float* hj_W1   = (const float*)d_in[10];
  const float* hj_b1   = (const float*)d_in[11];
  const float* hj_W2   = (const float*)d_in[12];
  const float* hj_b2   = (const float*)d_in[13];
  const float* ln_g    = (const float*)d_in[14];
  const float* ln_b    = (const float*)d_in[15];
  const float* Wq_lin  = (const float*)d_in[16];
  const float* Wk_lin  = (const float*)d_in[17];
  const float* gw_W1   = (const float*)d_in[18];
  const float* gw_b1   = (const float*)d_in[19];
  const float* gw_W2   = (const float*)d_in[20];
  const float* gw_b2   = (const float*)d_in[21];
  const float* gt_W1   = (const float*)d_in[22];
  const float* gt_b1   = (const float*)d_in[23];
  const float* gt_W2   = (const float*)d_in[24];
  const float* gt_b2   = (const float*)d_in[25];
  const float* W_query = (const float*)d_in[26];
  const float* W_key   = (const float*)d_in[27];
  const float* vu_W    = (const float*)d_in[28];
  const float* gm_W1   = (const float*)d_in[29];
  const float* gm_b1   = (const float*)d_in[30];
  const float* gm_W2   = (const float*)d_in[31];
  const float* gm_b2   = (const float*)d_in[32];

  float* out_h = (float*)d_out;
  float* out_X = out_h + N_NODESC * LATC;
  float* out_t = out_X + (size_t)N_NODESC * MULTC * SPHC;

  char* wp = (char*)d_ws;
  auto alloc = [&](size_t bytes) {
    void* p = (void*)wp;
    wp += (bytes + 255) & ~(size_t)255;
    return p;
  };
  unsigned short* wrs_frag = (unsigned short*)alloc((size_t)NTILES_ENV * 512 * 2);
  unsigned short* w2_frag  = (unsigned short*)alloc((size_t)NTILES_ENV * 512 * 2);
  unsigned short* wq_frag  = (unsigned short*)alloc((size_t)32 * 512 * 2);
  unsigned short* wk_frag  = (unsigned short*)alloc((size_t)32 * 512 * 2);
  float* XQ     = (float*)alloc(sizeof(float) * N_NODESC * MULTC * SPHC);
  float* XK     = (float*)alloc(sizeof(float) * N_NODESC * MULTC * SPHC);
  float* dh     = (float*)alloc(sizeof(float) * N_NODESC * LATC);
  float* h1     = (float*)alloc(sizeof(float) * N_NODESC * LATC);
  float* dXij   = (float*)alloc(sizeof(float) * (size_t)N_EDGESC * MULTC * SPHC);
  float* logits = (float*)alloc(sizeof(float) * N_EDGESC * MULTC);
  float* ewb    = (float*)alloc(sizeof(float) * N_EDGESC * MULTC);
  unsigned* smax= (unsigned*)alloc(sizeof(unsigned) * N_NODESC * MULTC);
  float* denom  = (float*)alloc(sizeof(float) * N_NODESC * MULTC);
  float* dXacc  = (float*)alloc(sizeof(float) * N_NODESC * MULTC * SPHC);

  hipMemsetAsync(dh, 0, sizeof(float) * N_NODESC * LATC, stream);
  hipMemsetAsync(smax, 0, sizeof(unsigned) * N_NODESC * MULTC, stream);
  hipMemsetAsync(denom, 0, sizeof(float) * N_NODESC * MULTC, stream);
  hipMemsetAsync(dXacc, 0, sizeof(float) * N_NODESC * MULTC * SPHC, stream);

  k_pack<<<(NTILES_ENV * 512 + 255) / 256, 256, 0, stream>>>(W_rs, wrs_frag, GEN_NC, NTILES_ENV);
  k_pack<<<(NTILES_ENV * 512 + 255) / 256, 256, 0, stream>>>(hj_W2, w2_frag, GEN_NC, NTILES_ENV);
  k_pack<<<(32 * 512 + 255) / 256, 256, 0, stream>>>(W_query, wq_frag, 512, 32);
  k_pack<<<(32 * 512 + 255) / 256, 256, 0, stream>>>(W_key, wk_frag, 512, 32);

  int nirr = N_NODESC * MULTC * SPHC;
  k_irreps<<<(nirr + 255) / 256, 256, 0, stream>>>(X, Wq_lin, XQ);
  k_irreps<<<(nirr + 255) / 256, 256, 0, stream>>>(X, Wk_lin, XK);

  k_env<<<N_EDGESC / EB, 256, 0, stream>>>(t_ij, h, X, ec, en, sph, hj_W1, hj_b1, hj_b2,
                                           wrs_frag, w2_frag, dh, dXij);
  k_dhln<<<N_NODESC, 32, 0, stream>>>(dh, h, ln_g, ln_b, h1);
  k_edge_attn<<<N_EDGESC / EB, 256, 0, stream>>>(t_ij, h, ec, en, XQ, XK,
      gw_W1, gw_b1, gw_W2, gw_b2, gt_W1, gt_b1, gt_W2, gt_b2, uc,
      wq_frag, wk_frag, out_t, logits, smax);
  k_softmax1<<<(N_EDGESC * MULTC + 255) / 256, 256, 0, stream>>>(logits, ec, smax, ewb, denom);
  k_scatterX<<<(N_EDGESC * MULTC + 255) / 256, 256, 0, stream>>>(ewb, denom, ec, dXij, dXacc);
  k_node_final<<<N_NODESC, 32, 0, stream>>>(X, h1, dXacc, vu_W, gm_W1, gm_b1, gm_W2, gm_b2,
                                            uc, out_h, out_X);
}
